// TrajectoryGRU_69982197121642
// MI455X (gfx1250) — matrix operand order repaired and checked by reading, not compiled, not run on hardware
//
#include <hip/hip_runtime.h>

// ---------------------------------------------------------------------------
// TrajectoryGRU on MI455X (gfx1250): persistent single-WGP kernel.
//   512 outer steps x 1024 inner GRU cells = 524,288 sequential cells.
//   Latency-bound serial chain -> keep ALL state on-chip:
//     - Whh (768x256) as bf16 WMMA B-fragments resident in VGPRs (192/lane)
//     - per-lane Wih row + biases as loop-invariant registers (13/lane)
//     - h (bf16 feed) in LDS, h (f32 master) in a per-lane register
//   Tile ownership: wave w owns gate tiles {w, w+16, w+32} == r/z/n rows for
//   hidden indices w*16..w*16+15, so each wave consumes its OWN accumulators
//   for the gate nonlinearity -> no gh LDS round-trip, ONE barrier per step.
//   A-matrix trick: only D row M=0 is consumed, so all lanes broadcast the
//   same h chunk; all D rows are duplicates, so every lane (incl. 16-31)
//   holds the correct matvec in acc[0] -> no EXEC predication anywhere in
//   the step.
// ---------------------------------------------------------------------------

typedef __attribute__((ext_vector_type(16))) __bf16 v16bf;
typedef __attribute__((ext_vector_type(8)))  float  v8f;

#define HSZ   256          // hidden size
#define GSZ   768          // 3*H (gates r,z,n)
#define TOBS  1024         // window length
#define DIN   3            // input size
#define NWAVE 16           // 16 wave32s
#define NTHR  (NWAVE * 32) // 512 threads
#define NTPW  3            // r/z/n tile per wave

static __device__ __forceinline__ unsigned short f2bfbits(float f) {
    // round-to-nearest-even f32 -> bf16 bit pattern
    unsigned u = __builtin_bit_cast(unsigned, f);
    return (unsigned short)((u + 0x7FFFu + ((u >> 16) & 1u)) >> 16);
}

static __device__ __forceinline__ float sigmoid_fast(float x) {
    return 1.0f / (1.0f + __expf(-x));          // v_exp_f32 + v_rcp path
}
static __device__ __forceinline__ float tanh_fast(float x) {
    // tanh(x) = 1 - 2/(1+e^{2x}); saturates correctly to +/-1 at extremes
    return 1.0f - 2.0f / (1.0f + __expf(2.0f * x));
}

__global__ void __launch_bounds__(NTHR, 1)
trajectory_gru_persistent(const float* __restrict__ xobs,   // (1,1024,3)
                          const float* __restrict__ Wih,    // (768,3)
                          const float* __restrict__ Whh,    // (768,256)
                          const float* __restrict__ bih,    // (768)
                          const float* __restrict__ bhh,    // (768)
                          const float* __restrict__ Wfc,    // (3,256)
                          const float* __restrict__ bfc,    // (3)
                          const int*   __restrict__ plen,   // scalar
                          float*       __restrict__ out)    // (3, pred_len)
{
    __shared__ float          s_xw[TOBS * 4];     // window ring, padded xyz0 -> b128
    __shared__ float          s_Wfc[DIN * HSZ];
    __shared__ float          s_bfc[4];
    __shared__ float          s_h[HSZ];           // f32 h, written once per outer iter
    __shared__ unsigned short s_hb[HSZ];          // h bf16 bits (WMMA feed)

    const int tid  = threadIdx.x;
    const int wave = tid >> 5;
    const int lane = tid & 31;

    // ------------------------ one-time init (trivial cost) ------------------
    for (int i = tid; i < TOBS; i += NTHR) {
        s_xw[i * 4 + 0] = xobs[i * 3 + 0];
        s_xw[i * 4 + 1] = xobs[i * 3 + 1];
        s_xw[i * 4 + 2] = xobs[i * 3 + 2];
        s_xw[i * 4 + 3] = 0.0f;
    }
    for (int i = tid; i < DIN * HSZ; i += NTHR) s_Wfc[i] = Wfc[i];
    if (tid < DIN) s_bfc[tid] = bfc[tid];
    if (tid < HSZ) s_hb[tid] = 0;                // torch h=None -> zeros

    // --- preload Whh^T tiles as bf16 WMMA B-fragments, resident in VGPRs ---
    // Wave w owns tiles {w, w+16, w+32}: the r/z/n gate rows for hidden
    // indices w*16..w*16+15. B (32x16 bf16) layout: lanes 0-15 hold K=0..7,16..23
    // of column N=lane, lanes 16-31 hold K=8..15,24..31 of column N=lane-16.
    // Column N of tile nt == row nt*16+N of Whh.
    v16bf Bf[NTPW][8];
    {
        const int col  = lane & 15;
        const int koff = (lane < 16) ? 0 : 8;
#pragma unroll
        for (int n = 0; n < NTPW; ++n) {
            const int row = (wave + n * NWAVE) * 16 + col;   // tiles w, w+16, w+32
#pragma unroll
            for (int k = 0; k < 8; ++k) {
                const float* p = Whh + row * HSZ + k * 32 + koff;
#pragma unroll
                for (int j = 0; j < 16; ++j)
                    Bf[n][k][j] = __builtin_bit_cast(__bf16, f2bfbits(p[j < 8 ? j : j + 8]));
            }
        }
    }

    // --- per-lane loop-invariant constants (13 VGPRs, loaded once) ---------
    // This lane's hidden element: lanes 16-31 mirror lanes 0-15 and compute
    // identical values redundantly (no EXEC manipulation anywhere).
    const int hidx = wave * 16 + (lane & 15);
    const int ir = hidx, iz = HSZ + hidx, in_ = 2 * HSZ + hidx;
    const float wr0 = Wih[ir * 3 + 0], wr1 = Wih[ir * 3 + 1], wr2 = Wih[ir * 3 + 2];
    const float wz0 = Wih[iz * 3 + 0], wz1 = Wih[iz * 3 + 1], wz2 = Wih[iz * 3 + 2];
    const float wn0 = Wih[in_ * 3 + 0], wn1 = Wih[in_ * 3 + 1], wn2 = Wih[in_ * 3 + 2];
    const float cbr = bih[ir] + bhh[ir];    // r-gate biases fold (outside sigmoid arg sum)
    const float cbz = bih[iz] + bhh[iz];    // z-gate biases fold
    const float bin = bih[in_];             // n-gate: bih outside r*(...), bhh inside
    const float bhn = bhh[in_];
    const int P = plen[0];

    // Broadcast A-row source: 16-bit A 16x32 layout puts K 0..7 / 16..23 in
    // lanes 0-15 and K 8..15 / 24..31 in lanes 16-31. All lanes in each half
    // read the SAME LDS address (broadcast, conflict-free); A rows 1..15
    // duplicate row 0, which is harmless since only D row 0 is consumed.
    const int abase = (lane & 16) ? 8 : 0;
    auto load_a = [&](int k) {
        const unsigned short* hp = s_hb + k * 32 + abase;
        v16bf a;
#pragma unroll
        for (int j = 0; j < 8; ++j) {
            a[j]     = __builtin_bit_cast(__bf16, hp[j]);        // K k*32+abase+0..7
            a[8 + j] = __builtin_bit_cast(__bf16, hp[16 + j]);   // K k*32+abase+16..23
        }
        return a;
    };

    float hf = 0.0f;   // f32 hidden-state master, register-resident

    __syncthreads();

    int head = 0;  // ring-buffer start (uniform across threads)
    for (int p = 0; p < P; ++p) {
        // ---- inner scan: re-encode full window from carried h ----
        for (int t = 0; t < TOBS; ++t) {
            // Input-gate part gi = Wih@x_t + biases: only x comes from LDS
            // (one b128 broadcast); FMAs use loop-invariant registers and
            // fill issue slots during the A-frag ds_load waits.
            const int   xi = ((head + t) & (TOBS - 1)) * 4;
            const float x0 = s_xw[xi], x1 = s_xw[xi + 1], x2 = s_xw[xi + 2];
            const float gir = cbr + wr0 * x0 + wr1 * x1 + wr2 * x2;
            const float giz = cbz + wz0 * x0 + wz1 * x1 + wz2 * x2;
            const float gin = bin + wn0 * x0 + wn1 * x1 + wn2 * x2;

            // ---- Whh @ h via 24 WMMAs (3 gate tiles x 8 K-tiles) ----
            v8f acc0 = {}, acc1 = {}, acc2 = {};
#pragma unroll
            for (int k = 0; k < 8; ++k) {
                const v16bf a = load_a(k);
                acc0 = __builtin_amdgcn_wmma_f32_16x16x32_bf16(false, a, false, Bf[0][k],
                                                               (short)0, acc0, false, false);
                acc1 = __builtin_amdgcn_wmma_f32_16x16x32_bf16(false, a, false, Bf[1][k],
                                                               (short)0, acc1, false, false);
                acc2 = __builtin_amdgcn_wmma_f32_16x16x32_bf16(false, a, false, Bf[2][k],
                                                               (short)0, acc2, false, false);
            }

            // ---- gate nonlinearity on OWN accumulators (no LDS round-trip).
            // All D rows are duplicates -> acc[0] is the correct matvec value
            // in every lane; lanes L and L+16 store the identical value to the
            // same s_hb address (well-defined, no predication).
            const float r = sigmoid_fast(gir + acc0[0]);
            const float z = sigmoid_fast(giz + acc1[0]);
            const float n = tanh_fast(gin + r * (acc2[0] + bhn));
            hf = (1.0f - z) * n + z * hf;
            s_hb[hidx] = f2bfbits(hf);
            __syncthreads();   // single barrier: h visible to all waves
        }

        // ---- fc head: next = Wfc @ h + bfc ; emit + slide window ----
        s_h[hidx] = hf;                          // publish f32 h once per outer iter
        __syncthreads();
        if (tid < DIN) {
            float acc = s_bfc[tid];
            for (int i = 0; i < HSZ; ++i) acc += s_Wfc[tid * HSZ + i] * s_h[i];
            out[tid * P + p]     = acc;          // preds.T -> shape (D, pred_len)
            s_xw[head * 4 + tid] = acc;          // drop oldest, append newest
        }
        __syncthreads();
        head = (head + 1) & (TOBS - 1);
    }
}

extern "C" void kernel_launch(void* const* d_in, const int* in_sizes, int n_in,
                              void* d_out, int out_size, void* d_ws, size_t ws_size,
                              hipStream_t stream) {
    (void)in_sizes; (void)n_in; (void)out_size; (void)d_ws; (void)ws_size;
    trajectory_gru_persistent<<<1, NTHR, 0, stream>>>(
        (const float*)d_in[0],   // observed_seq
        (const float*)d_in[1],   // Wih
        (const float*)d_in[2],   // Whh
        (const float*)d_in[3],   // bih
        (const float*)d_in[4],   // bhh
        (const float*)d_in[5],   // Wfc
        (const float*)d_in[6],   // bfc
        (const int*)  d_in[7],   // pred_len
        (float*)d_out);
}
